// FCOSPostProcessor_9345848836424
// MI455X (gfx1250) — compile-verified, matrix operand order checked by reading, not run on hardware
//
#include <hip/hip_runtime.h>

// ---------------- problem constants (match reference) ----------------
#define N_IMG   16
#define C_CLS   80
#define H_DIM   160
#define W_DIM   160
#define HW      (H_DIM * W_DIM)        // 25600
#define TOT     (HW * C_CLS)           // 2,048,000 per image
#define TOPK    1000
#define K_PAD   1024
#define CAP     2048
#define OUT_N   100
#define NBINS   1024
#define NMS_TH  0.6f
#define CLS_OFF 4096.0f
#define LOGIT_TH (-2.9444389792f)      // logit(0.05): sigmoid(x)>0.05 <=> x>this
#define CHUNK   2048                   // elements per block in streaming passes

typedef __attribute__((ext_vector_type(16))) __bf16 v16bf;
typedef __attribute__((ext_vector_type(8)))  float  v8f;

__device__ __forceinline__ float sigm(float x) { return 1.f / (1.f + __expf(-x)); }

// ---------------- K0: zero everything we read-before-write ----------------
__global__ void init_kernel(int* __restrict__ hist, int* __restrict__ counter,
                            unsigned* __restrict__ mask, float* __restrict__ out,
                            int out_n) {
    const int nh = N_IMG * NBINS;
    const int nc = N_IMG;
    const int nm = N_IMG * K_PAD * 32;
    const int total = nh + nc + nm + out_n;
    for (int i = blockIdx.x * blockDim.x + threadIdx.x; i < total;
         i += gridDim.x * blockDim.x) {
        if (i < nh)                hist[i] = 0;
        else if (i < nh + nc)      counter[i - nh] = 0;
        else if (i < nh + nc + nm) mask[i - nh - nc] = 0u;
        else                       out[i - nh - nc - nm] = 0.f;
    }
}

// ---------------- K1: streaming score histogram (pass 1 over box_cls) ------
__global__ void hist_kernel(const float* __restrict__ cls,
                            const float* __restrict__ ctr,
                            int* __restrict__ hist) {
    __shared__ int lh[NBINS];
    for (int t = threadIdx.x; t < NBINS; t += blockDim.x) lh[t] = 0;
    __syncthreads();
    const int bpi = TOT / CHUNK;                 // blocks per image
    const int n    = blockIdx.x / bpi;
    const int base = (blockIdx.x % bpi) * CHUNK;
    const float* clsN = cls + (size_t)n * TOT;
    const float* ctrN = ctr + (size_t)n * HW;
    for (int k = 0; k < CHUNK / 256; ++k) {
        int r = base + k * 256 + threadIdx.x;
        if (r + CHUNK < TOT) __builtin_prefetch(clsN + r + CHUNK, 0, 1);
        float x = clsN[r];
        if (x > LOGIT_TH) {                      // == sigmoid(x) > 0.05
            int   hw = r % HW;
            float s  = sigm(x) * sigm(ctrN[hw]);
            int bin = (int)(s * NBINS);
            bin = bin < 0 ? 0 : (bin > NBINS - 1 ? NBINS - 1 : bin);
            atomicAdd(&lh[bin], 1);
        }
    }
    __syncthreads();
    for (int t = threadIdx.x; t < NBINS; t += blockDim.x)
        if (lh[t]) atomicAdd(&hist[n * NBINS + t], lh[t]);
}

// ---------------- K2: per-image suffix scan -> top-1000 score threshold ----
__global__ void thresh_kernel(const int* __restrict__ hist, float* __restrict__ thresh) {
    __shared__ int s[NBINS];
    const int n = blockIdx.x, t = threadIdx.x;
    s[t] = hist[n * NBINS + t];
    __syncthreads();
    for (int off = 1; off < NBINS; off <<= 1) {
        int v = s[t] + ((t + off < NBINS) ? s[t + off] : 0);
        __syncthreads();
        s[t] = v;
        __syncthreads();
    }
    if (t == 0 && s[0] < TOPK) thresh[n] = 0.f;
    if (s[t] >= TOPK && (t == NBINS - 1 || s[t + 1] < TOPK))
        thresh[n] = (float)t / (float)NBINS;
}

// ---------------- K3: compaction (pass 2 over box_cls) ---------------------
__global__ void compact_kernel(const float* __restrict__ cls,
                               const float* __restrict__ ctr,
                               const float* __restrict__ thresh,
                               int* __restrict__ counter,
                               float* __restrict__ cscore,
                               int* __restrict__ cidx) {
    const int bpi = TOT / CHUNK;
    const int n    = blockIdx.x / bpi;
    const int base = (blockIdx.x % bpi) * CHUNK;
    const float th = thresh[n];
    const float* clsN = cls + (size_t)n * TOT;
    const float* ctrN = ctr + (size_t)n * HW;
    for (int k = 0; k < CHUNK / 256; ++k) {
        int r = base + k * 256 + threadIdx.x;
        float x = clsN[r];
        if (x > LOGIT_TH) {
            int   hw = r % HW;
            int   ch = r / HW;
            float s  = sigm(x) * sigm(ctrN[hw]);
            if (s >= th) {
                int pos = atomicAdd(&counter[n], 1);
                if (pos < CAP) {
                    cscore[n * CAP + pos] = s;
                    cidx[n * CAP + pos]   = hw * C_CLS + ch;  // flat (HW,C) index
                }
            }
        }
    }
}

// ---------------- K4: bitonic sort (desc by score, asc idx) + box decode ---
__global__ void sort_decode_kernel(const int* __restrict__ counter,
                                   const float* __restrict__ cscore,
                                   const int* __restrict__ cidx,
                                   const float* __restrict__ breg,
                                   const int* __restrict__ imsz,
                                   float* __restrict__ boxes,
                                   float* __restrict__ boxesOff,
                                   float* __restrict__ detScore,
                                   int* __restrict__ labels,
                                   float* __restrict__ areas) {
    __shared__ float ss[CAP];
    __shared__ int   si[CAP];
    const int n = blockIdx.x, tid = threadIdx.x;
    int cnt = counter[n];
    if (cnt > CAP) cnt = CAP;
    for (int e = tid; e < CAP; e += 1024) {
        if (e < cnt) { ss[e] = cscore[n * CAP + e]; si[e] = cidx[n * CAP + e]; }
        else         { ss[e] = -1.f;                si[e] = 0x7FFFFFFF; }
    }
    __syncthreads();
    for (int k = 2; k <= CAP; k <<= 1) {
        for (int j = k >> 1; j > 0; j >>= 1) {
            for (int e = tid; e < CAP; e += 1024) {
                int p = e ^ j;
                if (p > e) {
                    float s0 = ss[e], s1 = ss[p];
                    int   i0 = si[e], i1 = si[p];
                    bool lessE = (s0 < s1) || (s0 == s1 && i0 > i1);
                    bool up = ((e & k) == 0);              // descending segment
                    bool doSwap = up ? lessE : !lessE;
                    if (doSwap) { ss[e] = s1; ss[p] = s0; si[e] = i1; si[p] = i0; }
                }
            }
            __syncthreads();
        }
    }
    const float maxx = (float)imsz[n * 2 + 1] - 1.f;
    const float maxy = (float)imsz[n * 2 + 0] - 1.f;
    for (int t = tid; t < K_PAD; t += 1024) {
        float s = ss[t]; int id = si[t];
        bool valid = (t < TOPK) && (s > 0.f) && (id >= 0) && (id < TOT);
        float x1 = 0, y1 = 0, x2 = 0, y2 = 0, ds = 0; int lab = 0;
        if (valid) {
            int loc = id / C_CLS, ch = id % C_CLS;
            lab = ch + 1;
            float lx = (float)((loc % W_DIM) * 8 + 4);
            float ly = (float)((loc / W_DIM) * 8 + 4);
            const float* rg = breg + (size_t)n * 4 * HW;
            float r0 = rg[0 * HW + loc], r1 = rg[1 * HW + loc];
            float r2 = rg[2 * HW + loc], r3 = rg[3 * HW + loc];
            x1 = fminf(fmaxf(lx - r0, 0.f), maxx);
            x2 = fminf(fmaxf(lx + r2, 0.f), maxx);
            y1 = fminf(fmaxf(ly - r1, 0.f), maxy);
            y2 = fminf(fmaxf(ly + r3, 0.f), maxy);
            ds = sqrtf(fmaxf(s, 1e-12f));
        }
        size_t o = (size_t)n * K_PAD + t;
        boxes[o * 4 + 0] = x1; boxes[o * 4 + 1] = y1;
        boxes[o * 4 + 2] = x2; boxes[o * 4 + 3] = y2;
        float off = (float)lab * CLS_OFF;       // class-aware NMS offset
        boxesOff[o * 4 + 0] = x1 + off; boxesOff[o * 4 + 1] = y1 + off;
        boxesOff[o * 4 + 2] = x2 + off; boxesOff[o * 4 + 3] = y2 + off;
        detScore[o] = ds;
        labels[o]   = lab;
        areas[o]    = (x2 - x1) * (y2 - y1);
    }
}

// ---------------- K5: 16x16-tiled suppression bitmask; WMMA for area_i+area_j
__global__ void iou_mask_kernel(const float* __restrict__ boxesOff,
                                const float* __restrict__ areas,
                                unsigned* __restrict__ mask) {
    const int TPI = (K_PAD / 16) * (K_PAD / 16);   // 4096 tiles per image
    int wid  = blockIdx.x * (blockDim.x >> 5) + (threadIdx.x >> 5);
    int lane = threadIdx.x & 31;
    int n  = wid / TPI;
    int t  = wid % TPI;
    int ti = t / (K_PAD / 16), tj = t % (K_PAD / 16);
    int row0 = ti * 16, col0 = tj * 16;
    int m = lane & 15;

    // A = [area_i | 1 | 0...] (16x32), B = [1 ; area_j ; 0...] (32x16)
    // => C[M][N] = area_i[M] + area_j[N]  (rank-2 GEMM on the matrix core)
    v16bf A = {}; v16bf B = {};
    float areaI = areas[n * K_PAD + row0 + m];
    float areaJ = areas[n * K_PAD + col0 + m];
    if (lane < 16) {
        A[0] = (__bf16)areaI; A[1] = (__bf16)1.0f;
        B[0] = (__bf16)1.0f;  B[1] = (__bf16)areaJ;
    }
    v8f Cz = {};
    v8f S = __builtin_amdgcn_wmma_f32_16x16x32_bf16(false, A, false, B,
                                                    (short)0, Cz, false, false);

    int col = col0 + (lane & 15);
    const float4* bo = (const float4*)boxesOff;
    float4 bj = bo[(size_t)n * K_PAD + col];
    for (int r = 0; r < 8; ++r) {
        int row = row0 + r + ((lane >> 4) << 3);   // C layout: lanes 16-31 -> M+8
        float4 bi = bo[(size_t)n * K_PAD + row];
        float lx = fmaxf(bi.x, bj.x), ly = fmaxf(bi.y, bj.y);
        float rx = fminf(bi.z, bj.z), ry = fminf(bi.w, bj.w);
        float inter = fmaxf(rx - lx, 0.f) * fmaxf(ry - ly, 0.f);
        float uni   = S[r] - inter;                // area_i + area_j from WMMA
        // iou > TH  <=>  inter > TH * union  (both sides >= 0): avoids v_div chain
        bool bit = inter > NMS_TH * fmaxf(uni, 1e-9f);
#if __has_builtin(__builtin_amdgcn_ballot_w32)
        unsigned bal = __builtin_amdgcn_ballot_w32(bit);
        if (lane == 0) {
            int w = tj >> 1, sh = (tj & 1) * 16;
            atomicOr(&mask[(size_t)(n * K_PAD + row0 + r) * 32 + w],
                     (bal & 0xFFFFu) << sh);
            atomicOr(&mask[(size_t)(n * K_PAD + row0 + r + 8) * 32 + w],
                     (bal >> 16) << sh);
        }
#else
        if (bit) atomicOr(&mask[(size_t)(n * K_PAD + row) * 32 + (col >> 5)],
                          1u << (col & 31));
#endif
    }
}

// ---------------- K6: 1-wave greedy sweep over bitmask + top-100 output ----
__global__ void nms_out_kernel(const float* __restrict__ boxes,
                               const float* __restrict__ detScore,
                               const int* __restrict__ labels,
                               const unsigned* __restrict__ mask,
                               float* __restrict__ out) {
    __shared__ unsigned remv[32];
    __shared__ int keepIdx[OUT_N];
    __shared__ int kcount;
    __shared__ int kflag;
    const int n = blockIdx.x, lane = threadIdx.x;
    remv[lane] = 0u;
    if (lane == 0) kcount = 0;
    __syncthreads();
    for (int i = 0; i < TOPK; ++i) {
        if (lane == 0) {
            bool sup  = (remv[i >> 5] >> (i & 31)) & 1u;
            bool kept = !sup && (detScore[n * K_PAD + i] > 0.f) && (kcount < OUT_N);
            kflag = kept ? 1 : 0;
            if (kept) { keepIdx[kcount] = i; kcount = kcount + 1; }
        }
        __syncthreads();
        if (kflag) remv[lane] |= mask[(size_t)(n * K_PAD + i) * 32 + lane];
        __syncthreads();
    }
    float* oB = out;                               // [16,100,4]
    float* oS = out + N_IMG * OUT_N * 4;           // [16,100]
    float* oL = out + N_IMG * OUT_N * 5;           // [16,100]
    for (int t = lane; t < OUT_N; t += 32) {
        if (t < kcount) {
            int ci = keepIdx[t];
            size_t o = (size_t)n * K_PAD + ci;
            size_t q = (size_t)n * OUT_N + t;
            oB[q * 4 + 0] = boxes[o * 4 + 0];
            oB[q * 4 + 1] = boxes[o * 4 + 1];
            oB[q * 4 + 2] = boxes[o * 4 + 2];
            oB[q * 4 + 3] = boxes[o * 4 + 3];
            oS[q] = detScore[o];
            oL[q] = (float)labels[o];
        }
    }
}

// ---------------- host-side orchestration ----------------------------------
extern "C" void kernel_launch(void* const* d_in, const int* in_sizes, int n_in,
                              void* d_out, int out_size, void* d_ws, size_t ws_size,
                              hipStream_t stream) {
    (void)in_sizes; (void)n_in; (void)ws_size;
    // inputs: 0=locations (recomputed on device), 1=box_cls, 2=box_regression,
    //         3=centerness, 4=image_sizes
    const float* box_cls = (const float*)d_in[1];
    const float* box_reg = (const float*)d_in[2];
    const float* ctr     = (const float*)d_in[3];
    const int*   imsz    = (const int*)d_in[4];

    char* ws = (char*)d_ws;
    size_t off = 0;
    auto take = [&](size_t bytes) -> void* {
        off = (off + 255) & ~((size_t)255);
        void* p = ws + off;
        off += bytes;
        return p;
    };
    int*      hist     = (int*)     take((size_t)N_IMG * NBINS * 4);
    int*      counter  = (int*)     take((size_t)N_IMG * 4);
    float*    thresh   = (float*)   take((size_t)N_IMG * 4);
    float*    cscore   = (float*)   take((size_t)N_IMG * CAP * 4);
    int*      cidx     = (int*)     take((size_t)N_IMG * CAP * 4);
    float*    boxes    = (float*)   take((size_t)N_IMG * K_PAD * 16);
    float*    boxesOff = (float*)   take((size_t)N_IMG * K_PAD * 16);
    float*    detScore = (float*)   take((size_t)N_IMG * K_PAD * 4);
    int*      labels   = (int*)     take((size_t)N_IMG * K_PAD * 4);
    float*    areas    = (float*)   take((size_t)N_IMG * K_PAD * 4);
    unsigned* mask     = (unsigned*)take((size_t)N_IMG * K_PAD * 32 * 4);

    init_kernel<<<2048, 256, 0, stream>>>(hist, counter, mask, (float*)d_out, out_size);
    hist_kernel<<<N_IMG * (TOT / CHUNK), 256, 0, stream>>>(box_cls, ctr, hist);
    thresh_kernel<<<N_IMG, NBINS, 0, stream>>>(hist, thresh);
    compact_kernel<<<N_IMG * (TOT / CHUNK), 256, 0, stream>>>(box_cls, ctr, thresh,
                                                              counter, cscore, cidx);
    sort_decode_kernel<<<N_IMG, 1024, 0, stream>>>(counter, cscore, cidx, box_reg,
                                                   imsz, boxes, boxesOff, detScore,
                                                   labels, areas);
    iou_mask_kernel<<<(N_IMG * 4096) / 8, 256, 0, stream>>>(boxesOff, areas, mask);
    nms_out_kernel<<<N_IMG, 32, 0, stream>>>(boxes, detScore, labels, mask,
                                             (float*)d_out);
}
